// MultiHeadAttention_51470888075395
// MI455X (gfx1250) — matrix operand order repaired and checked by reading, not compiled, not run on hardware
//
#include <hip/hip_runtime.h>
#include <hip/hip_bf16.h>

// ---------------------------------------------------------------------------
// MHA for MI455X (gfx1250): bf16 WMMA pipeline, fp32 accumulation.
//   B=8, N=2048, E=768, H=8, D=96
// Kernels:
//   1) qkv_gemm : X @ {Wq,Wk,Wv} + b -> bf16 Q,K in [B,H,N,D], V^T in [B,H,D,N]
//   2) attn     : per-wave 16-row 2-pass streaming-softmax attention
//   3) out_gemm : O(bf16) @ Wo + bo -> fp32 d_out; A-tile staged to LDS by the
//                 Tensor Data Mover (tensor_load_to_lds, 6-arg clang-23 form).
// ---------------------------------------------------------------------------

typedef __bf16 v16bf __attribute__((ext_vector_type(16)));
typedef float  v8f   __attribute__((ext_vector_type(8)));
typedef unsigned int u32x4 __attribute__((ext_vector_type(4)));
typedef int          i32x4 __attribute__((ext_vector_type(4)));
typedef int          i32x8 __attribute__((ext_vector_type(8)));

#define BATCH 8
#define SEQ   2048
#define EMB   768
#define HEADS 8
#define HDIM  96

#if defined(__gfx1250__) && __has_builtin(__builtin_amdgcn_tensor_load_to_lds) && \
    __has_builtin(__builtin_amdgcn_s_wait_tensorcnt)
#define USE_TDM 1
#endif

union FragBF {
    v16bf v;
    unsigned short us[16];
    uint4 q[2];
};

__device__ __forceinline__ unsigned short f2bf(float f) {
    unsigned int u = __float_as_uint(f);
    unsigned int r = u + 0x7FFFu + ((u >> 16) & 1u);
    return (unsigned short)(r >> 16);
}

// Packed 2x f32 -> bf16 conversion (hardware v_cvt_pk_bf16_f32 when present).
__device__ __forceinline__ unsigned int f2bf_pk(float lo, float hi) {
#if defined(__gfx1250__) && __has_builtin(__builtin_amdgcn_cvt_pk_bf16_f32)
    auto v = __builtin_amdgcn_cvt_pk_bf16_f32(lo, hi);
    unsigned int u;
    __builtin_memcpy(&u, &v, 4);
    return u;
#else
    return (unsigned int)f2bf(lo) | ((unsigned int)f2bf(hi) << 16);
#endif
}

__device__ __forceinline__ v8f wmma_bf16(const FragBF& a, const FragBF& b, v8f c) {
    return __builtin_amdgcn_wmma_f32_16x16x32_bf16(false, a.v, false, b.v,
                                                   (short)0, c, false, false);
}

// ---------------------------------------------------------------------------
// Kernel 1: fused QKV projection. grid = (EMB/128, (B*N)/128, 3), block = 256.
// Each of 8 waves computes a 32x64 sub-tile of the 128x128 block tile.
// ---------------------------------------------------------------------------
__global__ void qkv_gemm(const float* __restrict__ x,
                         const float* __restrict__ Wq, const float* __restrict__ bq,
                         const float* __restrict__ Wk, const float* __restrict__ bk,
                         const float* __restrict__ Wv, const float* __restrict__ bv,
                         unsigned short* __restrict__ Q,
                         unsigned short* __restrict__ K,
                         unsigned short* __restrict__ Vt)
{
    __shared__ __align__(16) unsigned short ldsA[128][40]; // M x K (bf16)
    __shared__ __align__(16) unsigned short ldsB[128][40]; // N x K (transposed, bf16)

    const int tid  = threadIdx.x;
    const int lane = tid & 31;
    const int wave = tid >> 5;
    const int half = lane >> 4;
    const int l16  = lane & 15;
    const int wr   = wave >> 1;     // 0..3 -> 32-row strip
    const int wc   = wave & 1;      // 0..1 -> 64-col strip
    const int mb   = blockIdx.y * 128;
    const int nb   = blockIdx.x * 128;
    const int z    = blockIdx.z;

    const float* W    = (z == 0) ? Wq : ((z == 1) ? Wk : Wv);
    const float* bias = (z == 0) ? bq : ((z == 1) ? bk : bv);

    const v8f zero8 = {0.f, 0.f, 0.f, 0.f, 0.f, 0.f, 0.f, 0.f};
    v8f acc[2][4];
    for (int i = 0; i < 2; ++i)
        for (int j = 0; j < 4; ++j) acc[i][j] = zero8;

    for (int k0 = 0; k0 < EMB; k0 += 32) {
        // Stage A tile: 128x32 fp32 -> bf16 LDS (packed cvt, 2 dword stores)
        for (int l = 0; l < 4; ++l) {
            int idx = tid + l * 256;
            int row = idx >> 3;
            int cg  = (idx & 7) * 4;
            float4 v = *(const float4*)(x + (size_t)(mb + row) * EMB + k0 + cg);
            *(unsigned int*)&ldsA[row][cg]     = f2bf_pk(v.x, v.y);
            *(unsigned int*)&ldsA[row][cg + 2] = f2bf_pk(v.z, v.w);
        }
        // Stage B tile transposed: W[k][n] -> ldsB[n][k]
        for (int l = 0; l < 4; ++l) {
            int idx = tid + l * 256;
            int kr  = idx >> 5;
            int cg  = (idx & 31) * 4;
            float4 v = *(const float4*)(W + (size_t)(k0 + kr) * EMB + nb + cg);
            ldsB[cg + 0][kr] = f2bf(v.x);
            ldsB[cg + 1][kr] = f2bf(v.y);
            ldsB[cg + 2][kr] = f2bf(v.z);
            ldsB[cg + 3][kr] = f2bf(v.w);
        }
        __syncthreads();

        FragBF a[2], b[4];
        for (int rt = 0; rt < 2; ++rt) {
            const unsigned short* p = &ldsA[wr * 32 + rt * 16 + l16][16 * half];
            a[rt].q[0] = *(const uint4*)p;
            a[rt].q[1] = *(const uint4*)(p + 8);
        }
        for (int ct = 0; ct < 4; ++ct) {
            const unsigned short* p = &ldsB[wc * 64 + ct * 16 + l16][16 * half];
            b[ct].q[0] = *(const uint4*)p;
            b[ct].q[1] = *(const uint4*)(p + 8);
        }
        for (int rt = 0; rt < 2; ++rt)
            for (int ct = 0; ct < 4; ++ct)
                acc[rt][ct] = wmma_bf16(a[rt], b[ct], acc[rt][ct]);
        __syncthreads();
    }

    // Store with bias; scatter to head layouts.
    for (int rt = 0; rt < 2; ++rt) {
        for (int ct = 0; ct < 4; ++ct) {
            const int gc = nb + wc * 64 + ct * 16 + l16;
            const int h  = gc / HDIM;
            const int d  = gc % HDIM;
            const float bv_ = bias[gc];
            for (int r = 0; r < 8; ++r) {
                const int gr = mb + wr * 32 + rt * 16 + r + 8 * half;
                const int bb = gr >> 11;          // / SEQ
                const int n  = gr & (SEQ - 1);
                const unsigned short o = f2bf(acc[rt][ct][r] + bv_);
                if (z == 2) {
                    Vt[(((size_t)bb * HEADS + h) * HDIM + d) * SEQ + n] = o;
                } else {
                    unsigned short* dst = (z == 0) ? Q : K;
                    dst[(((size_t)bb * HEADS + h) * SEQ + n) * HDIM + d] = o;
                }
            }
        }
    }
}

// ---------------------------------------------------------------------------
// Kernel 2: attention. grid = (SEQ/64, B*H), block = 128 (4 waves).
// Each wave owns 16 query rows; 2-pass streaming softmax over 2048 keys.
// ---------------------------------------------------------------------------
__global__ void attn_kernel(const unsigned short* __restrict__ Qh,
                            const unsigned short* __restrict__ Kh,
                            const unsigned short* __restrict__ Vt,
                            unsigned short* __restrict__ Oa)
{
    __shared__ __align__(16) unsigned short Plds[4][16][32]; // per-wave P transpose

    const int tid  = threadIdx.x;
    const int wave = tid >> 5;
    const int lane = tid & 31;
    const int half = lane >> 4;
    const int l16  = lane & 15;
    const int bh   = blockIdx.y;
    const int qtb  = blockIdx.x * 64 + wave * 16;

    // Q A-fragments for the 3 K-steps of D=96 (kept in registers throughout)
    FragBF aQ[3];
    {
        const size_t qbase = ((size_t)bh * SEQ + qtb + l16) * HDIM;
        for (int s = 0; s < 3; ++s) {
            const unsigned short* p = Qh + qbase + s * 32 + 16 * half;
            aQ[s].q[0] = *(const uint4*)p;
            aQ[s].q[1] = *(const uint4*)(p + 8);
        }
    }

    auto computeS = [&](int j, v8f& s0, v8f& s1) {
        const v8f zero8 = {0.f, 0.f, 0.f, 0.f, 0.f, 0.f, 0.f, 0.f};
        s0 = zero8;
        s1 = zero8;
        const size_t kb0 = ((size_t)bh * SEQ + j + l16) * HDIM;
        const size_t kb1 = ((size_t)bh * SEQ + j + 16 + l16) * HDIM;
        for (int s = 0; s < 3; ++s) {
            FragBF b0, b1;
            const unsigned short* p0 = Kh + kb0 + s * 32 + 16 * half;
            const unsigned short* p1 = Kh + kb1 + s * 32 + 16 * half;
            b0.q[0] = *(const uint4*)p0;
            b0.q[1] = *(const uint4*)(p0 + 8);
            b1.q[0] = *(const uint4*)p1;
            b1.q[1] = *(const uint4*)(p1 + 8);
            s0 = wmma_bf16(aQ[s], b0, s0);
            s1 = wmma_bf16(aQ[s], b1, s1);
        }
    };

    // Pass 1: row maxima
    float mrow[8];
    for (int r = 0; r < 8; ++r) mrow[r] = -3.0e38f;
    for (int j = 0; j < SEQ; j += 32) {
        if (j + 32 < SEQ)
            __builtin_prefetch(Kh + ((size_t)bh * SEQ + j + 32 + l16) * HDIM, 0, 1);
        v8f s0, s1;
        computeS(j, s0, s1);
        for (int r = 0; r < 8; ++r)
            mrow[r] = fmaxf(mrow[r], fmaxf(s0[r], s1[r]));
    }
    for (int off = 1; off < 16; off <<= 1)
        for (int r = 0; r < 8; ++r)
            mrow[r] = fmaxf(mrow[r], __shfl_xor(mrow[r], off, 32));

    // Pass 2: exp, P*V accumulation (unnormalized)
    const v8f zero8 = {0.f, 0.f, 0.f, 0.f, 0.f, 0.f, 0.f, 0.f};
    float rsum[8] = {0.f, 0.f, 0.f, 0.f, 0.f, 0.f, 0.f, 0.f};
    v8f accO[6];
    for (int dt = 0; dt < 6; ++dt) accO[dt] = zero8;

    for (int j = 0; j < SEQ; j += 32) {
        v8f s0, s1;
        computeS(j, s0, s1);
        for (int r = 0; r < 8; ++r) {
            float p0 = __expf(s0[r] - mrow[r]);
            float p1 = __expf(s1[r] - mrow[r]);
            rsum[r] += p0 + p1;
            Plds[wave][r + 8 * half][l16]      = f2bf(p0);
            Plds[wave][r + 8 * half][16 + l16] = f2bf(p1);
        }
        // Transpose C-layout P into A-fragment layout via LDS (in-order per wave)
        FragBF aP;
        {
            const unsigned short* p = &Plds[wave][l16][16 * half];
            aP.q[0] = *(const uint4*)p;
            aP.q[1] = *(const uint4*)(p + 8);
        }
        for (int dt = 0; dt < 6; ++dt) {
            const unsigned short* vp =
                Vt + ((size_t)bh * HDIM + dt * 16 + l16) * SEQ + j + 16 * half;
            FragBF bV;
            bV.q[0] = *(const uint4*)vp;
            bV.q[1] = *(const uint4*)(vp + 8);
            accO[dt] = wmma_bf16(aP, bV, accO[dt]);
        }
    }

    for (int off = 1; off < 16; off <<= 1)
        for (int r = 0; r < 8; ++r)
            rsum[r] += __shfl_xor(rsum[r], off, 32);

    // out = (P*V) / (rowsum * sqrt(E))   [reference: softmax THEN /sqrt(E)]
    const float invSqrtE = 0.036084391824351615f; // 1/sqrt(768)
    const int b = bh >> 3;
    const int h = bh & 7;
    for (int r = 0; r < 8; ++r) {
        const float inv = invSqrtE / rsum[r];
        const int qr = qtb + r + 8 * half;
        const size_t rowbase = ((size_t)(b * SEQ + qr)) * EMB + h * HDIM;
        for (int dt = 0; dt < 6; ++dt)
            Oa[rowbase + dt * 16 + l16] = f2bf(accO[dt][r] * inv);
    }
}

// ---------------------------------------------------------------------------
// Kernel 3: output projection. grid = (EMB/128, (B*N)/128), block = 256.
// A tile (already bf16) is staged to LDS by the Tensor Data Mover; descriptor
// uses TDM padding (16 DW interval, 4 DW pad) to reproduce the ldsA[128][40]
// bank-padded layout. Uses the 6-arg clang-23 builtin form (extra i32x8 = 0).
// ---------------------------------------------------------------------------
__global__ void out_gemm(const unsigned short* __restrict__ Oa,
                         const float* __restrict__ Wo,
                         const float* __restrict__ bo,
                         float* __restrict__ out)
{
    __shared__ __align__(16) unsigned short ldsA[128][40];
    __shared__ __align__(16) unsigned short ldsB[128][40];

    const int tid  = threadIdx.x;
    const int lane = tid & 31;
    const int wave = tid >> 5;
    const int half = lane >> 4;
    const int l16  = lane & 15;
    const int wr   = wave >> 1;
    const int wc   = wave & 1;
    const int mb   = blockIdx.y * 128;
    const int nb   = blockIdx.x * 128;

    const v8f zero8 = {0.f, 0.f, 0.f, 0.f, 0.f, 0.f, 0.f, 0.f};
    v8f acc[2][4];
    for (int i = 0; i < 2; ++i)
        for (int j = 0; j < 4; ++j) acc[i][j] = zero8;

#if defined(USE_TDM)
    // Wave-uniform pieces of the tensor descriptor (D#).
    const unsigned int ldsA_base = (unsigned int)(size_t)&ldsA[0][0];
    const unsigned long long gbase =
        (unsigned long long)(size_t)Oa + (unsigned long long)mb * EMB * 2ull;
    // group1: data_size=2B (code 1), pad_enable, pad_interval=16 DW (code 3),
    //         pad_amount=4 DW (code 3); dims/tile per K-step below.
    i32x8 g1;
    g1[0] = (1 << 16) | (1 << 20) | (3 << 22) | (3 << 25);
    g1[1] = (int)((unsigned)(EMB & 0xFFFF) << 16);          // tensor_dim0 lo16
    g1[2] = (int)((EMB >> 16) | (((BATCH * SEQ) & 0xFFFF) << 16)); // dim0 hi / dim1 lo
    g1[3] = (int)(((BATCH * SEQ) >> 16) | (32 << 16));      // dim1 hi / tile_dim0=32
    g1[4] = 128;                                            // tile_dim1=128, tile_dim2=0
    g1[5] = EMB;                                            // tensor_dim0_stride lo32
    g1[6] = 0;
    g1[7] = 0;
    const i32x4 gz  = {0, 0, 0, 0};
    const i32x8 gz8 = {0, 0, 0, 0, 0, 0, 0, 0};
#endif

    for (int k0 = 0; k0 < EMB; k0 += 32) {
#if defined(USE_TDM)
        if (wave == 0) {
            const unsigned long long ga = gbase + (unsigned long long)k0 * 2ull;
            u32x4 g0;
            g0[0] = 1u;                                   // count=1 (user D#)
            g0[1] = ldsA_base;                            // lds_addr
            g0[2] = (unsigned int)(ga & 0xFFFFFFFFu);     // global_addr lo
            g0[3] = (unsigned int)((ga >> 32) & 0x01FFFFFFu) | (2u << 30); // hi | type=2
            __builtin_amdgcn_tensor_load_to_lds(g0, g1, gz, gz, gz8, 0);
        }
#else
        for (int l = 0; l < 4; ++l) {
            int idx = tid + l * 256;
            int row = idx >> 3;
            int cg  = (idx & 7) * 4;
            union { uint2 u; unsigned short s[4]; } t;
            t.u = *(const uint2*)(Oa + (size_t)(mb + row) * EMB + k0 + cg);
            ldsA[row][cg + 0] = t.s[0];
            ldsA[row][cg + 1] = t.s[1];
            ldsA[row][cg + 2] = t.s[2];
            ldsA[row][cg + 3] = t.s[3];
        }
#endif
        for (int l = 0; l < 4; ++l) {
            int idx = tid + l * 256;
            int kr  = idx >> 5;
            int cg  = (idx & 31) * 4;
            float4 v = *(const float4*)(Wo + (size_t)(k0 + kr) * EMB + nb + cg);
            ldsB[cg + 0][kr] = f2bf(v.x);
            ldsB[cg + 1][kr] = f2bf(v.y);
            ldsB[cg + 2][kr] = f2bf(v.z);
            ldsB[cg + 3][kr] = f2bf(v.w);
        }
#if defined(USE_TDM)
        if (wave == 0)
            __builtin_amdgcn_s_wait_tensorcnt(0);
#endif
        __syncthreads();

        FragBF a[2], b[4];
        for (int rt = 0; rt < 2; ++rt) {
            const unsigned short* p = &ldsA[wr * 32 + rt * 16 + l16][16 * half];
            a[rt].q[0] = *(const uint4*)p;
            a[rt].q[1] = *(const uint4*)(p + 8);
        }
        for (int ct = 0; ct < 4; ++ct) {
            const unsigned short* p = &ldsB[wc * 64 + ct * 16 + l16][16 * half];
            b[ct].q[0] = *(const uint4*)p;
            b[ct].q[1] = *(const uint4*)(p + 8);
        }
        for (int rt = 0; rt < 2; ++rt)
            for (int ct = 0; ct < 4; ++ct)
                acc[rt][ct] = wmma_bf16(a[rt], b[ct], acc[rt][ct]);
        __syncthreads();
    }

    for (int rt = 0; rt < 2; ++rt) {
        for (int ct = 0; ct < 4; ++ct) {
            const int gc = nb + wc * 64 + ct * 16 + l16;
            const float bv_ = bo[gc];
            for (int r = 0; r < 8; ++r) {
                const int gr = mb + wr * 32 + rt * 16 + r + 8 * half;
                out[(size_t)gr * EMB + gc] = acc[rt][ct][r] + bv_;
            }
        }
    }
}

// ---------------------------------------------------------------------------
extern "C" void kernel_launch(void* const* d_in, const int* in_sizes, int n_in,
                              void* d_out, int out_size, void* d_ws, size_t ws_size,
                              hipStream_t stream)
{
    const float* x  = (const float*)d_in[0];
    const float* Wq = (const float*)d_in[1];
    const float* bq = (const float*)d_in[2];
    const float* Wk = (const float*)d_in[3];
    const float* bk = (const float*)d_in[4];
    const float* Wv = (const float*)d_in[5];
    const float* bv = (const float*)d_in[6];
    const float* Wo = (const float*)d_in[7];
    const float* bo = (const float*)d_in[8];
    float* out = (float*)d_out;

    const size_t elems = (size_t)BATCH * SEQ * EMB; // 12,582,912
    unsigned short* Qh = (unsigned short*)d_ws;
    unsigned short* Kh = Qh + elems;
    unsigned short* Vt = Kh + elems;
    unsigned short* Oa = Vt + elems;

    qkv_gemm<<<dim3(EMB / 128, (BATCH * SEQ) / 128, 3), 256, 0, stream>>>(
        x, Wq, bq, Wk, bk, Wv, bv, Qh, Kh, Vt);

    attn_kernel<<<dim3(SEQ / 64, BATCH * HEADS), 128, 0, stream>>>(Qh, Kh, Vt, Oa);

    out_gemm<<<dim3(EMB / 128, (BATCH * SEQ) / 128), 256, 0, stream>>>(
        Oa, Wo, bo, out);
}
